// Model_5549097746586
// MI455X (gfx1250) — matrix operand order repaired and checked by reading, not compiled, not run on hardware
//
#include <hip/hip_runtime.h>
#include <hip/hip_bf16.h>

#define B_ 4096
#define T_ 256
#define F_ 20

typedef __bf16 bf16_t;
typedef bf16_t v16bf __attribute__((ext_vector_type(16)));
typedef float  v8f   __attribute__((ext_vector_type(8)));

union FragU { uint4 u[2]; v16bf v; };

__device__ __forceinline__ v8f wmma_bf16(v16bf a, v16bf b, v8f c) {
  return __builtin_amdgcn_wmma_f32_16x16x32_bf16(
      /*neg_a=*/false, a, /*neg_b=*/false, b,
      /*c_mod=*/(short)0, c, /*reuse_a=*/false, /*reuse_b=*/false);
}

// A fragment (16x32 bf16, M across lanes). ISA layout:
// lanes 0-15:  halves 0..7 = K k0+0..7,  halves 8..15 = K k0+16..23
// lanes 16-31: halves 0..7 = K k0+8..15, halves 8..15 = K k0+24..31
__device__ __forceinline__ v16bf ldsA(const bf16_t* buf, int strideH, int k0, int lane) {
  const bf16_t* p = buf + (lane & 15) * strideH + k0 + ((lane >> 4) << 3);
  FragU f;
  f.u[0] = *(const uint4*)(p);
  f.u[1] = *(const uint4*)(p + 16);
  return f.v;
}

// B fragment (32x16 bf16, N across lanes) from weights stored transposed [N][Kpad]:
// lanes 0-15: K k0+0..7,16..23 ; lanes 16-31: K k0+8..15,24..31 ; col = n0 + (lane&15)
__device__ __forceinline__ v16bf ldsB(const bf16_t* wt, int kpad, int n0, int k0, int lane) {
  const bf16_t* p = wt + (n0 + (lane & 15)) * kpad + k0 + ((lane >> 4) << 3);
  FragU f;
  f.u[0] = *(const uint4*)(p);
  f.u[1] = *(const uint4*)(p + 16);
  return f.v;
}

__device__ __forceinline__ float frcp(float x)  { return __builtin_amdgcn_rcpf(x); }
__device__ __forceinline__ float sigm(float x)  { return frcp(1.f + __expf(-x)); }
__device__ __forceinline__ float tanha(float x) { float e = __expf(2.f * x); return (e - 1.f) * frcp(e + 1.f); }

// Gather-transpose f32 [K][N] global weights -> bf16 [N][Kpad] LDS (zero padded)
__device__ __forceinline__ void loadWT(bf16_t* dst, const float* __restrict__ src,
                                       int K, int N, int kshift, int tid) {
  const int kpad  = 1 << kshift;
  const int total = N << kshift;
  for (int idx = tid; idx < total; idx += 64) {
    int n = idx >> kshift;
    int k = idx & (kpad - 1);
    float v = (k < K) ? src[k * N + n] : 0.f;
    dst[idx] = (bf16_t)v;
  }
}

// One GRU layer step for a 16-row batch tile.
// NT = 3*U/16 gate tiles total, KX = input K-steps (of 32), KH = hidden K-steps.
// WMMA chains start from inline-zero C; biases are folded into the elementwise
// stage as per-lane scalars (bias is constant along M, varies along N = lane&15).
template<int NT, int KX, int KH>
__device__ __forceinline__ void gru_layer(const bf16_t* xbuf, bf16_t* hbuf,
                                          const bf16_t* kt, const bf16_t* rkt,
                                          const float* bx, const float* bh,
                                          v8f* hst, int lane) {
  constexpr int XS = KX * 32;
  constexpr int HS = KH * 32;
  constexpr int UT = NT / 3;
  const int nl = lane & 15;
  v16bf ax[KX], ah[KH];
#pragma unroll
  for (int s = 0; s < KX; ++s) ax[s] = ldsA(xbuf, XS, 32 * s, lane);
#pragma unroll
  for (int s = 0; s < KH; ++s) ah[s] = ldsA(hbuf, HS, 32 * s, lane);
#pragma unroll
  for (int j = 0; j < UT; ++j) {
    float bz  = bx[16 * j + nl]            + bh[16 * j + nl];
    float br  = bx[16 * (UT + j) + nl]     + bh[16 * (UT + j) + nl];
    float bxh = bx[16 * (2 * UT + j) + nl];
    float bhh = bh[16 * (2 * UT + j) + nl];
    v8f xz = {}, xr = {}, xh = {}, hz = {}, hr = {}, hh = {};
#pragma unroll
    for (int s = 0; s < KX; ++s) {
      xz = wmma_bf16(ax[s], ldsB(kt, XS, 16 * j,            32 * s, lane), xz);
      xr = wmma_bf16(ax[s], ldsB(kt, XS, 16 * (UT + j),     32 * s, lane), xr);
      xh = wmma_bf16(ax[s], ldsB(kt, XS, 16 * (2 * UT + j), 32 * s, lane), xh);
    }
#pragma unroll
    for (int s = 0; s < KH; ++s) {
      hz = wmma_bf16(ah[s], ldsB(rkt, HS, 16 * j,            32 * s, lane), hz);
      hr = wmma_bf16(ah[s], ldsB(rkt, HS, 16 * (UT + j),     32 * s, lane), hr);
      hh = wmma_bf16(ah[s], ldsB(rkt, HS, 16 * (2 * UT + j), 32 * s, lane), hh);
    }
    v8f hn;
#pragma unroll
    for (int v = 0; v < 8; ++v) {
      float z  = sigm(xz[v] + hz[v] + bz);
      float r  = sigm(xr[v] + hr[v] + br);
      float hc = tanha((xh[v] + bxh) + r * (hh[v] + bhh));
      hn[v] = hc + z * (hst[j][v] - hc);
    }
    hst[j] = hn;
    // D-layout -> row-major bf16 in LDS (transpose via strided 16-bit stores)
    int n  = 16 * j + nl;
    int mb = (lane >> 4) << 3;
#pragma unroll
    for (int v = 0; v < 8; ++v) hbuf[(mb + v) * HS + n] = (bf16_t)hn[v];
  }
}

__global__ __launch_bounds__(64, 2)
void gru4_kernel(const float* __restrict__ in, const float* __restrict__ Wemb,
                 const float* __restrict__ k0, const float* __restrict__ rk0, const float* __restrict__ b0,
                 const float* __restrict__ k1, const float* __restrict__ rk1, const float* __restrict__ b1,
                 const float* __restrict__ k2, const float* __restrict__ rk2, const float* __restrict__ b2,
                 const float* __restrict__ k3, const float* __restrict__ rk3, const float* __restrict__ b3,
                 const float* __restrict__ Wd, const float* __restrict__ bd,
                 float* __restrict__ out) {
  // Weights transposed [N][Kpad] bf16, shared by both waves of the block
  __shared__ __align__(16) bf16_t s_k0t [192 * 32];
  __shared__ __align__(16) bf16_t s_rk0t[192 * 64];
  __shared__ __align__(16) bf16_t s_k1t [96 * 64];
  __shared__ __align__(16) bf16_t s_rk1t[96 * 32];
  __shared__ __align__(16) bf16_t s_k2t [96 * 32];
  __shared__ __align__(16) bf16_t s_rk2t[96 * 32];
  __shared__ __align__(16) bf16_t s_k3t [96 * 32];
  __shared__ __align__(16) bf16_t s_rk3t[96 * 32];
  __shared__ __align__(16) float  s_b0[2 * 192];
  __shared__ __align__(16) float  s_b1[2 * 96];
  __shared__ __align__(16) float  s_b2[2 * 96];
  __shared__ __align__(16) float  s_b3[2 * 96];
  // Per-wave activation staging, row-major [16][Kpad]
  __shared__ __align__(16) bf16_t s_x [2][16 * 32];
  __shared__ __align__(16) bf16_t s_h0[2][16 * 64];
  __shared__ __align__(16) bf16_t s_h1[2][16 * 32];
  __shared__ __align__(16) bf16_t s_h2[2][16 * 32];
  __shared__ __align__(16) bf16_t s_h3[2][16 * 32];

  const int tid  = threadIdx.x;
  const int wv   = tid >> 5;
  const int lane = tid & 31;
  const int bt   = blockIdx.x * 32 + wv * 16;   // this wave's batch tile

  // Cooperative one-time staging (both waves)
  loadWT(s_k0t,  k0,  23, 192, 5, tid);
  loadWT(s_rk0t, rk0, 64, 192, 6, tid);
  loadWT(s_k1t,  k1,  64,  96, 6, tid);
  loadWT(s_rk1t, rk1, 32,  96, 5, tid);
  loadWT(s_k2t,  k2,  32,  96, 5, tid);
  loadWT(s_rk2t, rk2, 32,  96, 5, tid);
  loadWT(s_k3t,  k3,  32,  96, 5, tid);
  loadWT(s_rk3t, rk3, 32,  96, 5, tid);
  for (int i = tid; i < 2 * 192; i += 64) s_b0[i] = b0[i];
  for (int i = tid; i < 2 * 96; i += 64) {
    s_b1[i] = b1[i]; s_b2[i] = b2[i]; s_b3[i] = b3[i];
  }
  // Zero staging (h init = 0; X pad cols stay 0 forever)
  for (int i = tid; i < 2 * 16 * 32; i += 64) {
    s_x[0][i]  = (bf16_t)0.f;
    s_h1[0][i] = (bf16_t)0.f;
    s_h2[0][i] = (bf16_t)0.f;
    s_h3[0][i] = (bf16_t)0.f;
  }
  for (int i = tid; i < 2 * 16 * 64; i += 64) s_h0[0][i] = (bf16_t)0.f;

  __syncthreads();   // only barrier: weights/bias image ready; waves run free after

  const bf16_t* xw  = s_x[wv];
  bf16_t* h0w = s_h0[wv];
  bf16_t* h1w = s_h1[wv];
  bf16_t* h2w = s_h2[wv];
  bf16_t* h3w = s_h3[wv];

  // Persistent hidden-state accumulators (WMMA C/D layout)
  v8f h0s[4] = {}, h1s[2] = {}, h2s[2] = {}, h3s[2] = {};

  for (int t = 0; t < T_; ++t) {
    // Build X tile [16 rows x 32 cols bf16]: [f0, in[2..19], emb(4), pad]
    if (lane < 16) {
      const float* p = in + ((size_t)(bt + lane) * T_ + (size_t)t) * F_;
      float4 a0 = *(const float4*)(p);
      float4 a1 = *(const float4*)(p + 4);
      float4 a2 = *(const float4*)(p + 8);
      float4 a3 = *(const float4*)(p + 12);
      float4 a4 = *(const float4*)(p + 16);
      int id = (int)a0.y;
      float4 e = *(const float4*)(Wemb + 4 * (size_t)id);
      bf16_t* xr = s_x[wv] + lane * 32;
      xr[0]  = (bf16_t)a0.x;
      xr[1]  = (bf16_t)a0.z;  xr[2]  = (bf16_t)a0.w;
      xr[3]  = (bf16_t)a1.x;  xr[4]  = (bf16_t)a1.y;  xr[5]  = (bf16_t)a1.z;  xr[6]  = (bf16_t)a1.w;
      xr[7]  = (bf16_t)a2.x;  xr[8]  = (bf16_t)a2.y;  xr[9]  = (bf16_t)a2.z;  xr[10] = (bf16_t)a2.w;
      xr[11] = (bf16_t)a3.x;  xr[12] = (bf16_t)a3.y;  xr[13] = (bf16_t)a3.z;  xr[14] = (bf16_t)a3.w;
      xr[15] = (bf16_t)a4.x;  xr[16] = (bf16_t)a4.y;  xr[17] = (bf16_t)a4.z;  xr[18] = (bf16_t)a4.w;
      xr[19] = (bf16_t)e.x;   xr[20] = (bf16_t)e.y;   xr[21] = (bf16_t)e.z;   xr[22] = (bf16_t)e.w;
    }
    // Per-wave recurrence; LDS ops are in-order within a wave, no barrier needed.
    gru_layer<12, 1, 2>(xw,  h0w, s_k0t, s_rk0t, s_b0, s_b0 + 192, h0s, lane);
    gru_layer<6,  2, 1>(h0w, h1w, s_k1t, s_rk1t, s_b1, s_b1 + 96,  h1s, lane);
    gru_layer<6,  1, 1>(h1w, h2w, s_k2t, s_rk2t, s_b2, s_b2 + 96,  h2s, lane);
    gru_layer<6,  1, 1>(h2w, h3w, s_k3t, s_rk3t, s_b3, s_b3 + 96,  h3s, lane);
  }

  // Dense 32->2 + softmax from the final h3 tile in LDS
  if (lane < 16) {
    float a0 = bd[0], a1 = bd[1];
    const bf16_t* hr = h3w + lane * 32;
#pragma unroll
    for (int k = 0; k < 32; ++k) {
      float hv = (float)hr[k];
      a0 += hv * Wd[2 * k];
      a1 += hv * Wd[2 * k + 1];
    }
    float m  = fmaxf(a0, a1);
    float e0 = __expf(a0 - m), e1 = __expf(a1 - m);
    float inv = frcp(e0 + e1);
    out[2 * (size_t)(bt + lane)]     = e0 * inv;
    out[2 * (size_t)(bt + lane) + 1] = e1 * inv;
  }
}

extern "C" void kernel_launch(void* const* d_in, const int* in_sizes, int n_in,
                              void* d_out, int out_size, void* d_ws, size_t ws_size,
                              hipStream_t stream) {
  (void)in_sizes; (void)n_in; (void)out_size; (void)d_ws; (void)ws_size;
  const float* in   = (const float*)d_in[0];
  const float* Wemb = (const float*)d_in[1];
  const float* k0   = (const float*)d_in[2];
  const float* rk0  = (const float*)d_in[3];
  const float* b0   = (const float*)d_in[4];
  const float* k1   = (const float*)d_in[5];
  const float* rk1  = (const float*)d_in[6];
  const float* b1   = (const float*)d_in[7];
  const float* k2   = (const float*)d_in[8];
  const float* rk2  = (const float*)d_in[9];
  const float* b2   = (const float*)d_in[10];
  const float* k3   = (const float*)d_in[11];
  const float* rk3  = (const float*)d_in[12];
  const float* b3   = (const float*)d_in[13];
  const float* Wd   = (const float*)d_in[14];
  const float* bd   = (const float*)d_in[15];
  float* out = (float*)d_out;

  dim3 grid(B_ / 32);   // 128 blocks, 2 waves each: two independent 16-row tiles
  dim3 block(64);       // 2 x wave32 sharing the LDS weight image
  gru4_kernel<<<grid, block, 0, stream>>>(in, Wemb, k0, rk0, b0, k1, rk1, b1,
                                          k2, rk2, b2, k3, rk3, b3, Wd, bd, out);
}
